// RPNModel_84920093376984
// MI455X (gfx1250) — matrix operand order repaired and checked by reading, not compiled, not run on hardware
//
#include <hip/hip_runtime.h>

// ---------------------------------------------------------------------------
// RPN (conv3x3+ReLU -> cls/bbox heads -> softmax objectness -> decode -> top512)
// for MI455X / gfx1250. Conv + heads via v_wmma_f32_16x16x32_f16 (wave32).
// ---------------------------------------------------------------------------

typedef __attribute__((ext_vector_type(16))) _Float16 v16h;
typedef __attribute__((ext_vector_type(8)))  float    v8f;

union Frag16 { v16h v; uint4 q[2]; };

__device__ __forceinline__ v8f wmma16(v16h a, v16h b, v8f c) {
  // 8 args: (neg_a, A, neg_b, B, c_mod, C, reuse_a, reuse_b)
  return __builtin_amdgcn_wmma_f32_16x16x32_f16(false, a, false, b, (short)0, c,
                                                false, false);
}

#define HH 192
#define WW 192
#define CC 128
#define FF 128
#define BB 8
#define KA 9
#define NANCH (HH * WW * KA)   // 331776 anchors per batch
#define NROI 512
#define TOPCAP 4096

// ---------------------------------------------------------------------------
// Kernel 0: weight preprocessing.
//   wconvW[tap][n][k] = f16(w_conv[kh][kw][k][n])      (B-fragment friendly)
//   headwW[n][k]: n 0..17 = w_cls, n 32..67 = w_bbox, zero-padded to 80 rows
//   headbW[n]   : matching biases
// ---------------------------------------------------------------------------
__global__ void rpn_prep(const float* __restrict__ wconv_f,
                         const float* __restrict__ wcls,
                         const float* __restrict__ bcls,
                         const float* __restrict__ wbbox,
                         const float* __restrict__ bbbox,
                         _Float16* __restrict__ wconvW,
                         _Float16* __restrict__ headwW,
                         float* __restrict__ headbW) {
  int i = blockIdx.x * blockDim.x + threadIdx.x;
  const int NW = 9 * 128 * 128;
  if (i < NW) {
    int tap = i >> 14;            // kh*3+kw
    int n = (i >> 7) & 127;       // out channel
    int k = i & 127;              // in channel
    wconvW[i] = (_Float16)wconv_f[(tap * 128 + k) * 128 + n];
  } else if (i < NW + 80 * 128) {
    int j = i - NW;
    int n = j >> 7, k = j & 127;
    float v = 0.f;
    if (n < 18)                 v = wcls[k * 18 + n];
    else if (n >= 32 && n < 68) v = wbbox[k * 36 + (n - 32)];
    headwW[j] = (_Float16)v;
  } else if (i < NW + 80 * 128 + 80) {
    int n = i - NW - 80 * 128;
    float v = 0.f;
    if (n < 18)                 v = bcls[n];
    else if (n >= 32 && n < 68) v = bbbox[n - 32];
    headbW[n] = v;
  }
}

// ---------------------------------------------------------------------------
// Kernel 1: fused conv3x3(relu) + 1x1 heads + objectness/decode.
// One 256-thread (8-wave) workgroup per (batch, row, 32-pixel strip).
// Implicit GEMM: M=32 pixels, N=128 channels, K=9*128; each wave owns 16x32.
// ---------------------------------------------------------------------------
__global__ __launch_bounds__(256)
void rpn_conv_heads(const float* __restrict__ feat,
                    const float* __restrict__ bconv,
                    const _Float16* __restrict__ wconv,
                    const _Float16* __restrict__ headw,
                    const float* __restrict__ headb,
                    float* __restrict__ objW,
                    float* __restrict__ boxW) {
  __shared__ __align__(16) _Float16 in_rows[3][34][128];  // 26112 B halo
  __shared__ __align__(16) _Float16 xbuf[32][128];        //  8192 B conv out
  __shared__ float headbuf[32][80];                       // 10240 B head out

  const int tid = threadIdx.x;
  const int w0 = blockIdx.x * 32;
  const int h  = blockIdx.y;
  const int bz = blockIdx.z;

  // ---- stage input halo (3 rows x 34 cols x 128 ch) into LDS as f16 ----
  for (int i = tid; i < 3 * 34 * 128; i += 256) {
    int cpos = i & 127;
    int jj = (i >> 7) % 34;
    int rr = (i >> 7) / 34;
    int row = h + rr - 1, col = w0 + jj - 1;
    float v = 0.f;
    if (row >= 0 && row < HH && col >= 0 && col < WW)
      v = feat[(((size_t)bz * HH + row) * WW + col) * CC + cpos];
    in_rows[rr][jj][cpos] = (_Float16)v;
  }
  __syncthreads();

  const int wv = tid >> 5, lane = tid & 31;
  const int hs = lane >> 4;            // half-wave select
  const int l15 = lane & 15;           // M (for A) / N (for B,C)
  const int mBase = (wv & 1) * 16;     // pixel tile base
  const int nBase = (wv >> 1) * 32;    // channel tile base

  v8f acc0 = {0.f, 0.f, 0.f, 0.f, 0.f, 0.f, 0.f, 0.f};
  v8f acc1 = {0.f, 0.f, 0.f, 0.f, 0.f, 0.f, 0.f, 0.f};

  // ---- conv main loop: 9 taps x 4 K-chunks of 32 ----
  for (int tap = 0; tap < 9; ++tap) {
    const int kh = tap / 3, kw = tap % 3;
    const _Float16* w0p = wconv + (tap * 128 + nBase + l15) * 128;
    const _Float16* w1p = w0p + 16 * 128;
    const _Float16* ar = &in_rows[kh][mBase + l15 + kw][0];
#pragma unroll
    for (int c = 0; c < 4; ++c) {
      const int k0 = c * 32;
      Frag16 a, b0, b1;
      // A 16x32 f16 layout: lane half 0 -> K 0..7 & 16..23, half 1 -> 8..15 & 24..31
      a.q[0] = *(const uint4*)(ar + k0 + hs * 8);
      a.q[1] = *(const uint4*)(ar + k0 + 16 + hs * 8);
      // B 32x16 f16 layout: lane half 0 -> K 0..7 & 16..23, half 1 -> 8..15 & 24..31
      b0.q[0] = *(const uint4*)(w0p + k0 + hs * 8);
      b0.q[1] = *(const uint4*)(w0p + k0 + 16 + hs * 8);
      b1.q[0] = *(const uint4*)(w1p + k0 + hs * 8);
      b1.q[1] = *(const uint4*)(w1p + k0 + 16 + hs * 8);
      acc0 = wmma16(a.v, b0.v, acc0);
      acc1 = wmma16(a.v, b1.v, acc1);
    }
  }

  // ---- epilogue: bias + relu -> xbuf (f16) ----
  {
    const float bias0 = bconv[nBase + l15];
    const float bias1 = bconv[nBase + 16 + l15];
#pragma unroll
    for (int r = 0; r < 8; ++r) {
      const int mm = mBase + r + 8 * hs;   // C layout: VGPR r, lane half -> M
      float v0 = acc0[r] + bias0; v0 = v0 > 0.f ? v0 : 0.f;
      float v1 = acc1[r] + bias1; v1 = v1 > 0.f ? v1 : 0.f;
      xbuf[mm][nBase + l15] = (_Float16)v0;
      xbuf[mm][nBase + 16 + l15] = (_Float16)v1;
    }
  }
  __syncthreads();

  // ---- head GEMM: M=32 x N=80(padded) x K=128, 10 tiles over 8 waves ----
  for (int t = wv; t < 10; t += 8) {
    const int mT = t / 5, nT = t % 5;
    v8f hacc = {0.f, 0.f, 0.f, 0.f, 0.f, 0.f, 0.f, 0.f};
    const _Float16* hw = headw + (nT * 16 + l15) * 128;
    const _Float16* ar = &xbuf[mT * 16 + l15][0];
#pragma unroll
    for (int c = 0; c < 4; ++c) {
      const int k0 = c * 32;
      Frag16 a, b;
      a.q[0] = *(const uint4*)(ar + k0 + hs * 8);
      a.q[1] = *(const uint4*)(ar + k0 + 16 + hs * 8);
      b.q[0] = *(const uint4*)(hw + k0 + hs * 8);
      b.q[1] = *(const uint4*)(hw + k0 + 16 + hs * 8);
      hacc = wmma16(a.v, b.v, hacc);
    }
    const float hb = headb[nT * 16 + l15];
#pragma unroll
    for (int r = 0; r < 8; ++r)
      headbuf[mT * 16 + r + 8 * hs][nT * 16 + l15] = hacc[r] + hb;
  }
  __syncthreads();

  // ---- objectness + validity + box decode for 32 pixels x 9 anchors ----
  for (int ii = tid; ii < 32 * KA; ii += 256) {
    const int p = ii / KA, a = ii % KA;
    const int wcol = w0 + p;
    const float WIN[3] = {2.f, 4.f, 8.f};
    const float wwa = WIN[a % 3], hha = WIN[a / 3];
    const float l0 = headbuf[p][a];
    const float l1 = headbuf[p][9 + a];
    float obj = 1.f / (1.f + expf(l0 - l1));   // softmax class-1 prob
    const float xf = (float)wcol, yf = (float)h;
    const bool vx = (xf - wwa * 0.5f >= 0.f) && (xf + wwa * 0.5f <= (float)(WW - 1));
    const bool vy = (yf - hha * 0.5f >= 0.f) && (yf + hha * 0.5f <= (float)(HH - 1));
    if (!(vx && vy)) obj = 0.f;
    const float d0 = headbuf[p][32 + a];
    const float d1 = headbuf[p][32 + 9 + a];
    float d2 = headbuf[p][32 + 18 + a];
    float d3 = headbuf[p][32 + 27 + a];
    d2 = fminf(fmaxf(d2, -10.f), 10.f);
    d3 = fminf(fmaxf(d3, -10.f), 10.f);
    const float ox = xf - d0 * wwa;
    const float oy = yf - d1 * hha;
    const float ow = wwa * expf(d2);
    const float oh = hha * expf(d3);
    const size_t aidx = (size_t)bz * NANCH + ((size_t)h * WW + wcol) * KA + a;
    objW[aidx] = obj;
    float4 bx = {ox, oy, ow, oh};
    *(float4*)&boxW[aidx * 4] = bx;
  }
}

// ---------------------------------------------------------------------------
// Kernel 2: exact per-batch top-512 (descending obj, stable by index).
// One 1024-thread block per batch: 16-bit radix histogram -> cutoff bin ->
// candidate compaction -> in-LDS bitonic sort of u64 (rkey,idx) -> gather.
// ---------------------------------------------------------------------------
__global__ __launch_bounds__(1024)
void rpn_topk(const float* __restrict__ objW,
              const float* __restrict__ boxW,
              unsigned int* __restrict__ histW,
              float* __restrict__ out) {
  __shared__ unsigned long long cand[TOPCAP];  // 32 KB
  __shared__ unsigned int partial[1024];
  __shared__ unsigned int s_cnt, s_cut;

  const int tid = threadIdx.x;
  const int bz = blockIdx.x;
  unsigned int* hist = histW + (size_t)bz * 65536;
  const float* obj = objW + (size_t)bz * NANCH;

  // phase 1: clear histogram
  for (int i = tid; i < 65536; i += 1024) hist[i] = 0u;
  __threadfence();
  __syncthreads();

  // phase 2: histogram of top 16 bits of rkey (~sortable(obj) => ascending
  // rkey == descending objectness)
  for (int i = tid; i < NANCH; i += 1024) {
    unsigned int bits = __float_as_uint(obj[i]);
    unsigned int s = bits ^ ((bits >> 31) ? 0xFFFFFFFFu : 0x80000000u);
    unsigned int rkey = ~s;
    atomicAdd(&hist[rkey >> 16], 1u);
  }
  __threadfence();
  __syncthreads();

  // phase 3: find cutoff bin (first bin where cumulative count >= 512)
  {
    unsigned int sum = 0;
    for (int j = 0; j < 64; ++j) sum += hist[tid * 64 + j];
    partial[tid] = sum;
  }
  __syncthreads();
  if (tid == 0) {
    unsigned int acc = 0, cut = 65535u;
    int g = 0;
    while (g < 1024 && acc + partial[g] < NROI) { acc += partial[g]; ++g; }
    if (g < 1024) {
      for (int j = 0; j < 64; ++j) {
        unsigned int c = hist[g * 64 + j];
        if (acc + c >= NROI) { cut = (unsigned int)(g * 64 + j); break; }
        acc += c;
      }
    }
    s_cut = cut;
    s_cnt = 0u;
  }
  __syncthreads();
  const unsigned int cut = s_cut;

  // phase 4: compact candidates (bin <= cutoff) into LDS
  for (int i = tid; i < NANCH; i += 1024) {
    unsigned int bits = __float_as_uint(obj[i]);
    unsigned int s = bits ^ ((bits >> 31) ? 0xFFFFFFFFu : 0x80000000u);
    unsigned int rkey = ~s;
    if ((rkey >> 16) <= cut) {
      unsigned int pos = atomicAdd(&s_cnt, 1u);
      if (pos < TOPCAP)
        cand[pos] = ((unsigned long long)rkey << 32) | (unsigned int)i;
    }
  }
  __syncthreads();
  unsigned int M = s_cnt;
  if (M > TOPCAP) M = TOPCAP;
  for (int i = (int)M + tid; i < TOPCAP; i += 1024) cand[i] = ~0ull;
  __syncthreads();

  // phase 5: bitonic sort (ascending); full u64 key => deterministic order
  for (int k = 2; k <= TOPCAP; k <<= 1) {
    for (int j = k >> 1; j > 0; j >>= 1) {
      for (int t = tid; t < TOPCAP; t += 1024) {
        const int ixj = t ^ j;
        if (ixj > t) {
          unsigned long long va = cand[t], vb = cand[ixj];
          const bool up = ((t & k) == 0);
          if ((va > vb) == up) { cand[t] = vb; cand[ixj] = va; }
        }
      }
      __syncthreads();
    }
  }

  // phase 6: gather boxes for first 512 (sorted by descending objectness)
  for (int i = tid; i < NROI; i += 1024) {
    unsigned int idx = (unsigned int)(cand[i] & 0xFFFFFFFFu);
    if (idx >= (unsigned int)NANCH) idx = 0;
    const float4 bx = *(const float4*)&boxW[((size_t)bz * NANCH + idx) * 4];
    *(float4*)&out[((size_t)bz * NROI + i) * 4] = bx;
  }
}

// ---------------------------------------------------------------------------
// Launcher. Workspace layout (bytes):
//   [0)        wconv f16  9*128*128*2   = 294912
//   [294912)   headw f16  80*128*2      = 20480
//   [315392)   headb f32  80*4          = 320
//   [315904)   obj   f32  8*331776*4    = 10616832
//   [10932736) boxes f32  8*331776*4*4  = 42467328
//   [53400064) hist  u32  8*65536*4     = 2097152   (total ~53 MB)
// ---------------------------------------------------------------------------
extern "C" void kernel_launch(void* const* d_in, const int* in_sizes, int n_in,
                              void* d_out, int out_size, void* d_ws, size_t ws_size,
                              hipStream_t stream) {
  (void)in_sizes; (void)n_in; (void)out_size; (void)ws_size;

  const float* feat   = (const float*)d_in[0];
  const float* wconvf = (const float*)d_in[1];
  const float* bconv  = (const float*)d_in[2];
  const float* wcls   = (const float*)d_in[3];
  const float* bcls   = (const float*)d_in[4];
  const float* wbbox  = (const float*)d_in[5];
  const float* bbbox  = (const float*)d_in[6];

  char* ws = (char*)d_ws;
  _Float16* wconvW = (_Float16*)(ws + 0);
  _Float16* headwW = (_Float16*)(ws + 294912);
  float*    headbW = (float*)(ws + 315392);
  float*    objW   = (float*)(ws + 315904);
  float*    boxW   = (float*)(ws + 315904 + (size_t)BB * NANCH * 4);
  unsigned int* histW =
      (unsigned int*)(ws + 315904 + (size_t)BB * NANCH * 4 + (size_t)BB * NANCH * 16);

  const int prepN = 9 * 128 * 128 + 80 * 128 + 80;
  rpn_prep<<<(prepN + 255) / 256, 256, 0, stream>>>(
      wconvf, wcls, bcls, wbbox, bbbox, wconvW, headwW, headbW);

  dim3 g1(WW / 32, HH, BB);
  rpn_conv_heads<<<g1, 256, 0, stream>>>(
      feat, bconv, wconvW, headwW, headbW, objW, boxW);

  rpn_topk<<<BB, 1024, 0, stream>>>(objW, boxW, histW, (float*)d_out);
}
